// SimplifiedSparseAttention1D_16045997817878
// MI455X (gfx1250) — matrix operand order repaired and checked by reading, not compiled, not run on hardware
//
#include <hip/hip_runtime.h>
#include <stdint.h>

#define LLEN 4096
#define BATCH 4
#define CHN 64
#define CQ 16
#define NHASH 4

typedef __attribute__((ext_vector_type(16))) __bf16 v16bf;
typedef __attribute__((ext_vector_type(8)))  float  v8f;

union BV {
  v16bf v;
  uint4 q[2];
};

__device__ __forceinline__ unsigned short f2bf(float f) {
  unsigned int u = __float_as_uint(f);
  u = (u + 0x7FFFu + ((u >> 16) & 1u)) >> 16;   // round-to-nearest-even
  return (unsigned short)u;
}

// ---------------------------------------------------------------------------
// Kernel 1: q/k/v projections (fp32 compute, bf16 output) + hash-code packing.
//   q -> [b][l][16] bf16, scaled by log2(e) (so scores live in the exp2 domain)
//   k -> [b][l][16] bf16
//   v -> [b][ch][l] bf16   (channel-major: contiguous B-operand rows for P*V)
//   codes -> [b][l] u32, 4 hash ids packed one per byte
// ---------------------------------------------------------------------------
__global__ __launch_bounds__(256) void k1_qkv(
    const float* __restrict__ x, const int* __restrict__ hidx,
    const float* __restrict__ Wq, const float* __restrict__ bq,
    const float* __restrict__ Wk, const float* __restrict__ bk,
    const float* __restrict__ Wv, const float* __restrict__ bv,
    unsigned short* __restrict__ qg, unsigned short* __restrict__ kg,
    unsigned short* __restrict__ vg, unsigned int* __restrict__ codes)
{
  __shared__ float xs[64][65];
  __shared__ float wv[64][65];
  __shared__ float wq[16][65];
  __shared__ float wk[16][65];
  const int tid = threadIdx.x;
  const int b = blockIdx.y;
  const int pbase = blockIdx.x * 64;

  for (int i = tid; i < 64 * 64; i += 256) wv[i >> 6][i & 63] = Wv[i];
  for (int i = tid; i < 16 * 64; i += 256) { wq[i >> 6][i & 63] = Wq[i]; wk[i >> 6][i & 63] = Wk[i]; }
  for (int i = tid; i < 64 * 64; i += 256) {
    int c = i >> 6, p = i & 63;
    xs[c][p] = x[((size_t)b * CHN + c) * LLEN + pbase + p];
  }
  __syncthreads();

  { // v: thread -> (channel, 16 positions)
    const int co = tid & 63, pg = tid >> 6;
    for (int i = 0; i < 16; ++i) {
      const int pos = pg * 16 + i;
      float acc = bv[co];
#pragma unroll
      for (int ci = 0; ci < 64; ++ci) acc = fmaf(wv[co][ci], xs[ci][pos], acc);
      vg[((size_t)b * CHN + co) * LLEN + pbase + pos] = f2bf(acc);
    }
  }
  { // q,k: thread -> (cq, 4 positions)
    const int cq = tid & 15, pg = tid >> 4;
    for (int i = 0; i < 4; ++i) {
      const int pos = pg * 4 + i;
      float aq = bq[cq], ak = bk[cq];
#pragma unroll
      for (int ci = 0; ci < 64; ++ci) {
        const float xv = xs[ci][pos];
        aq = fmaf(wq[cq][ci], xv, aq);
        ak = fmaf(wk[cq][ci], xv, ak);
      }
      qg[((size_t)b * LLEN + pbase + pos) * CQ + cq] = f2bf(aq * 1.4426950408889634f);
      kg[((size_t)b * LLEN + pbase + pos) * CQ + cq] = f2bf(ak);
    }
  }
  if (tid < 64) {
    const int pos = tid;
    unsigned int c = 0;
#pragma unroll
    for (int h = 0; h < NHASH; ++h) {
      unsigned int v = (unsigned int)hidx[((size_t)b * NHASH + h) * LLEN + pbase + pos] & 255u;
      c |= v << (8 * h);
    }
    codes[(size_t)b * LLEN + pbase + pos] = c;
  }
}

// ---------------------------------------------------------------------------
// Kernel 2: fused multi-hash masked attention, flash-style, two passes.
// One wave owns a 16-row query tile; 4 waves / workgroup.
// WMMA D layout: reg r holds rows r (lanes 0-15) / r+8 (lanes 16-31), col=lane&15.
// ---------------------------------------------------------------------------
__global__ __launch_bounds__(128) void k2_attn(
    const unsigned short* __restrict__ qg,
    const unsigned short* __restrict__ kg,
    const unsigned short* __restrict__ vg,
    const unsigned int* __restrict__ codes,
    float* __restrict__ ao)   // [b][l][64] fp32
{
  __shared__ __align__(16) unsigned short sP[4][16][32];  // per-wave P bounce
  const int t    = threadIdx.x & 31;
  const int wave = threadIdx.x >> 5;
  const int b    = blockIdx.y;
  const int rowbase = (blockIdx.x * 4 + wave) * 16;
  const int half = t >> 4;
  const int lo   = t & 15;
  const size_t kbase = (size_t)b * LLEN;

  // Q tile as WMMA A operand: lane m: halfs0-15 = ch0..15 of row m;
  // lanes 16-31 (K=16..31) zero-padded.
  BV aq = {};
  if (half == 0) { aq.q[0] = *(const uint4*)(qg + (kbase + rowbase + lo) * CQ); aq.q[1] = *(const uint4*)(qg + (kbase + rowbase + lo) * CQ + 8); }

  unsigned int rcode[8];
#pragma unroll
  for (int r = 0; r < 8; ++r)
    rcode[r] = codes[kbase + rowbase + r + 8 * half];

  float m[8];
  float sm[NHASH][8];
#pragma unroll
  for (int r = 0; r < 8; ++r) {
    m[r] = -1e30f;
#pragma unroll
    for (int h = 0; h < NHASH; ++h) sm[h][r] = 0.f;
  }

  // ---------------- pass 1: global row max + per-hash denominators ----------------
  for (int cb = 0; cb < LLEN; cb += 32) {
    BV bk0 = {}, bk1 = {};
    if (t < 16) {  // B operand: lane n = key col, halfs = 16 channels; K=16..31 lanes zero
      const uint4* kp0 = (const uint4*)(kg + (kbase + cb + t) * CQ);
      bk0.q[0] = kp0[0]; bk0.q[1] = kp0[1];
      const uint4* kp1 = (const uint4*)(kg + (kbase + cb + 16 + t) * CQ);
      bk1.q[0] = kp1[0]; bk1.q[1] = kp1[1];
    }
    v8f z = {};
    v8f s0 = __builtin_amdgcn_wmma_f32_16x16x32_bf16(false, aq.v, false, bk0.v, (short)0, z, false, false);
    v8f s1 = __builtin_amdgcn_wmma_f32_16x16x32_bf16(false, aq.v, false, bk1.v, (short)0, z, false, false);

    float nm[8]; bool grew = false;
#pragma unroll
    for (int r = 0; r < 8; ++r) {
      nm[r] = fmaxf(m[r], fmaxf(s0[r], s1[r]));
      grew = grew || (nm[r] > m[r]);
    }
    if (__builtin_amdgcn_ballot_w32(grew)) {  // uniform skip when max unchanged
#pragma unroll
      for (int r = 0; r < 8; ++r) {
        const float sc = exp2f(m[r] - nm[r]);
#pragma unroll
        for (int h = 0; h < NHASH; ++h) sm[h][r] *= sc;
      }
    }
#pragma unroll
    for (int r = 0; r < 8; ++r) m[r] = nm[r];

    const unsigned int cc0 = codes[kbase + cb + lo];
    const unsigned int cc1 = codes[kbase + cb + 16 + lo];
#pragma unroll
    for (int r = 0; r < 8; ++r) {
      const float e0 = exp2f(s0[r] - m[r]);
      const float e1 = exp2f(s1[r] - m[r]);
      const unsigned int x0 = rcode[r] ^ cc0;
      const unsigned int x1 = rcode[r] ^ cc1;
#pragma unroll
      for (int h = 0; h < NHASH; ++h) {
        const float a = (((x0 >> (8 * h)) & 255u) == 0u) ? e0 : 0.f;
        const float c = (((x1 >> (8 * h)) & 255u) == 0u) ? e1 : 0.f;
        sm[h][r] += a + c;
      }
    }
  }

  // finalize: reduce max/sums across the 16 lanes holding each row; fold 1/NHASH
#pragma unroll
  for (int r = 0; r < 8; ++r) {
    float mr = m[r];
#pragma unroll
    for (int d = 1; d < 16; d <<= 1) mr = fmaxf(mr, __shfl_xor(mr, d, 32));
    const float sc = exp2f(m[r] - mr);
    m[r] = mr;
#pragma unroll
    for (int h = 0; h < NHASH; ++h) {
      float s = sm[h][r] * sc;
#pragma unroll
      for (int d = 1; d < 16; d <<= 1) s += __shfl_xor(s, d, 32);
      sm[h][r] = 0.25f / s;
    }
  }

  // ---------------- pass 2: combined-probability P*V via WMMA ----------------
  v8f acc[4];
#pragma unroll
  for (int n = 0; n < 4; ++n) acc[n] = (v8f){};

  for (int cb = 0; cb < LLEN; cb += 32) {
    BV bk0 = {}, bk1 = {};
    if (t < 16) {
      const uint4* kp0 = (const uint4*)(kg + (kbase + cb + t) * CQ);
      bk0.q[0] = kp0[0]; bk0.q[1] = kp0[1];
      const uint4* kp1 = (const uint4*)(kg + (kbase + cb + 16 + t) * CQ);
      bk1.q[0] = kp1[0]; bk1.q[1] = kp1[1];
    }
    v8f z = {};
    v8f s0 = __builtin_amdgcn_wmma_f32_16x16x32_bf16(false, aq.v, false, bk0.v, (short)0, z, false, false);
    v8f s1 = __builtin_amdgcn_wmma_f32_16x16x32_bf16(false, aq.v, false, bk1.v, (short)0, z, false, false);

    const unsigned int cc0 = codes[kbase + cb + lo];
    const unsigned int cc1 = codes[kbase + cb + 16 + lo];
#pragma unroll
    for (int r = 0; r < 8; ++r) {
      const float e0 = exp2f(s0[r] - m[r]);
      const float e1 = exp2f(s1[r] - m[r]);
      const unsigned int x0 = rcode[r] ^ cc0;
      const unsigned int x1 = rcode[r] ^ cc1;
      float w0 = 0.f, w1 = 0.f;
#pragma unroll
      for (int h = 0; h < NHASH; ++h) {
        const float iv = sm[h][r];
        w0 += (((x0 >> (8 * h)) & 255u) == 0u) ? iv : 0.f;
        w1 += (((x1 >> (8 * h)) & 255u) == 0u) ? iv : 0.f;
      }
      sP[wave][r + 8 * half][lo]      = f2bf(e0 * w0);
      sP[wave][r + 8 * half][lo + 16] = f2bf(e1 * w1);
    }
    asm volatile("s_wait_dscnt 0x0" ::: "memory");

    BV ap;  // reload P in A-operand layout (row-major LDS, contiguous b128 reads)
    {
      const unsigned short* rowp = &sP[wave][lo][0];
      ap.q[0] = *(const uint4*)(rowp + half * 16);
      ap.q[1] = *(const uint4*)(rowp + half * 16 + 8);
    }

#pragma unroll
    for (int n = 0; n < 4; ++n) {  // B operand: lane = channel, halfs = 16 positions
      BV bvt;
      const uint4* vp = (const uint4*)(vg + ((size_t)b * CHN + n * 16 + lo) * LLEN + cb + half * 16);
      bvt.q[0] = vp[0]; bvt.q[1] = vp[1];
      acc[n] = __builtin_amdgcn_wmma_f32_16x16x32_bf16(false, ap.v, false, bvt.v, (short)0, acc[n], false, false);
    }
  }

#pragma unroll
  for (int n = 0; n < 4; ++n)
#pragma unroll
    for (int r = 0; r < 8; ++r)
      ao[((size_t)b * LLEN + rowbase + r + 8 * half) * CHN + n * 16 + lo] = acc[n][r];
}

// ---------------------------------------------------------------------------
// Kernel 3a: out-proj + residual; per-workgroup deterministic BN partials.
// ---------------------------------------------------------------------------
__global__ __launch_bounds__(256) void k3a(
    const float* __restrict__ ao, const float* __restrict__ x,
    const float* __restrict__ Wo, const float* __restrict__ bo,
    const float* __restrict__ gamma,
    float* __restrict__ y, float* __restrict__ partial)
{
  __shared__ float wo[64][65];
  __shared__ float as[64][65];
  __shared__ float red[2][4][64];
  const int tid = threadIdx.x;
  const int b = blockIdx.y;
  const int pbase = blockIdx.x * 64;

  for (int i = tid; i < 64 * 64; i += 256) wo[i >> 6][i & 63] = Wo[i];
  for (int i = tid; i < 64 * 64; i += 256) as[i >> 6][i & 63] = ao[((size_t)b * LLEN + pbase) * CHN + i];
  __syncthreads();

  const float g = gamma[0];
  const int co = tid & 63, pg = tid >> 6;
  float s1 = 0.f, s2 = 0.f;
  for (int i = 0; i < 16; ++i) {
    const int pos = pg * 16 + i;
    float acc = bo[co];
#pragma unroll
    for (int ci = 0; ci < 64; ++ci) acc = fmaf(wo[co][ci], as[pos][ci], acc);
    const float o = g * acc + x[((size_t)b * CHN + co) * LLEN + pbase + pos];
    y[((size_t)b * CHN + co) * LLEN + pbase + pos] = o;
    s1 += o; s2 += o * o;
  }
  red[0][pg][co] = s1;
  red[1][pg][co] = s2;
  __syncthreads();
  if (tid < 64) {
    float a = 0.f, q = 0.f;
#pragma unroll
    for (int p = 0; p < 4; ++p) { a += red[0][p][tid]; q += red[1][p][tid]; }
    const int wg = blockIdx.y * gridDim.x + blockIdx.x;
    partial[(size_t)wg * 128 + tid] = a;
    partial[(size_t)wg * 128 + 64 + tid] = q;
  }
}

__global__ void k3b(const float* __restrict__ partial,
                    const float* __restrict__ bn_w, const float* __restrict__ bn_b,
                    float* __restrict__ ss, int nwg)
{
  const int c = threadIdx.x;
  if (c >= 64) return;
  float a = 0.f, q = 0.f;
  for (int w = 0; w < nwg; ++w) { a += partial[(size_t)w * 128 + c]; q += partial[(size_t)w * 128 + 64 + c]; }
  const float inv = 1.0f / (float)(BATCH * LLEN);
  const float mean = a * inv;
  const float var  = q * inv - mean * mean;
  const float rstd = rsqrtf(var + 1e-5f);
  const float sc = rstd * bn_w[c];
  ss[c] = sc;
  ss[64 + c] = bn_b[c] - mean * sc;
}

__global__ __launch_bounds__(256) void k3c(const float* __restrict__ y,
                                           const float* __restrict__ ss,
                                           float* __restrict__ out)
{
  const size_t i = (size_t)blockIdx.x * 256 + threadIdx.x;
  const int c = (int)((i >> 12) & 63);  // L = 4096
  out[i] = y[i] * ss[c] + ss[64 + c];
}

// ---------------------------------------------------------------------------
extern "C" void kernel_launch(void* const* d_in, const int* in_sizes, int n_in,
                              void* d_out, int out_size, void* d_ws, size_t ws_size,
                              hipStream_t stream) {
  (void)in_sizes; (void)n_in; (void)out_size; (void)ws_size;
  const float* x    = (const float*)d_in[0];
  const int*   hid  = (const int*)d_in[1];
  const float* Wq   = (const float*)d_in[2];
  const float* bq   = (const float*)d_in[3];
  const float* Wk   = (const float*)d_in[4];
  const float* bk   = (const float*)d_in[5];
  const float* Wv   = (const float*)d_in[6];
  const float* bv   = (const float*)d_in[7];
  const float* Wo   = (const float*)d_in[8];
  const float* bo   = (const float*)d_in[9];
  const float* gam  = (const float*)d_in[10];
  const float* bnw  = (const float*)d_in[11];
  const float* bnb  = (const float*)d_in[12];
  float* out = (float*)d_out;

  char* ws = (char*)d_ws;
  size_t off = 0;
  auto take = [&](size_t bytes) { char* p = ws + off; off = (off + bytes + 255) & ~(size_t)255; return p; };
  unsigned short* qg    = (unsigned short*)take((size_t)BATCH * LLEN * CQ * 2);
  unsigned short* kg    = (unsigned short*)take((size_t)BATCH * LLEN * CQ * 2);
  unsigned short* vg    = (unsigned short*)take((size_t)BATCH * CHN * LLEN * 2);
  unsigned int*   codes = (unsigned int*)take((size_t)BATCH * LLEN * 4);
  float*          ao    = (float*)take((size_t)BATCH * LLEN * CHN * 4);
  float*          y     = (float*)take((size_t)BATCH * CHN * LLEN * 4);
  const int nwg3 = (LLEN / 64) * BATCH;
  float*          part  = (float*)take((size_t)nwg3 * 128 * 4);
  float*          ss    = (float*)take(128 * 4);

  dim3 g1(LLEN / 64, BATCH);
  k1_qkv<<<g1, 256, 0, stream>>>(x, hid, Wq, bq, Wk, bk, Wv, bv, qg, kg, vg, codes);

  dim3 g2(LLEN / 64, BATCH);
  k2_attn<<<g2, 128, 0, stream>>>(qg, kg, vg, codes, ao);

  dim3 g3(LLEN / 64, BATCH);
  k3a<<<g3, 256, 0, stream>>>(ao, x, Wo, bo, gam, y, part);
  k3b<<<1, 64, 0, stream>>>(part, bnw, bnb, ss, nwg3);
  k3c<<<(BATCH * CHN * LLEN) / 256, 256, 0, stream>>>(y, ss, out);
}
